// EfficientMultiheadAttention_19937238188251
// MI455X (gfx1250) — matrix operand order repaired and checked by reading, not compiled, not run on hardware
//
#include <hip/hip_runtime.h>

// ---------------------------------------------------------------------------
// CDNA5 (gfx1250) multi-head attention, bf16 WMMA pipeline, wave32.
// sched_group_barrier pins a [loads][WMMA][loads][WMMA] software pipeline in
// the GEMM k-loop so the scheduler cannot sink loads onto their consumers
// (which previously forced register reuse and s_wait_loadcnt 0x0 stalls).
// ---------------------------------------------------------------------------

typedef __attribute__((ext_vector_type(16))) __bf16       v16bf;
typedef __attribute__((ext_vector_type(8)))  float        v8f;
typedef __attribute__((ext_vector_type(4)))  unsigned int u32x4;
typedef __attribute__((ext_vector_type(4)))  float        f32x4;
typedef __attribute__((ext_vector_type(4)))  __bf16       v4bf;

union AF { v16bf v; u32x4 q[2]; };   // 16x32 bf16 A-frag or 32x16 B-frag (8 VGPRs)
union CF { v8f v; float e[8]; };     // 16x16 f32 C/D frag (8 VGPRs)

static constexpr int Bdim = 2;
static constexpr int T    = 2048;
static constexpr int C    = 2048;
static constexpr int H    = 16;
static constexpr int Dh   = 128;
static constexpr int M    = Bdim * T;   // 4096 rows
static constexpr int N1   = 3 * C;      // 6144 qkv cols
static constexpr float QSCALE = 0.08838834764831845f;  // 1/sqrt(128)

// sched_group_barrier masks
#define SG_WMMA      0x008
#define SG_VMEM_READ 0x020
#define SG_DS_READ   0x100
#define SG_DS_WRITE  0x200

static __device__ __forceinline__ v8f wmma_bf16(v16bf a, v16bf b, v8f c) {
  return __builtin_amdgcn_wmma_f32_16x16x32_bf16(
      false, a, false, b, (short)0, c, false, false);
}

// A-frag: lane holds row m=lane&15; chunks at k + 8*hi and k + 16 + 8*hi.
static __device__ __forceinline__ AF load_afrag(const __bf16* p) {
  AF f;
  f.q[0] = *reinterpret_cast<const u32x4*>(p);
  f.q[1] = *reinterpret_cast<const u32x4*>(p + 16);
  return f;
}
// B-frag: lane holds col n=lane&15; chunks at k + 8*hi and k + 16 + 8*hi.
static __device__ __forceinline__ AF load_bfrag(const __bf16* p) {
  AF f;
  f.q[0] = *reinterpret_cast<const u32x4*>(p);
  f.q[1] = *reinterpret_cast<const u32x4*>(p + 16);
  return f;
}

// ---------------------------------------------------------------------------
// Kernel 1: fp32 -> bf16 cast (vectorized)
// ---------------------------------------------------------------------------
__global__ void __launch_bounds__(256) cast_f32_bf16(const float* __restrict__ src,
                                                     __bf16* __restrict__ dst, int n4) {
  int i = blockIdx.x * blockDim.x + threadIdx.x;
  if (i >= n4) return;
  f32x4 v = reinterpret_cast<const f32x4*>(src)[i];
  v4bf o;
  o[0] = (__bf16)v[0]; o[1] = (__bf16)v[1];
  o[2] = (__bf16)v[2]; o[3] = (__bf16)v[3];
  reinterpret_cast<v4bf*>(dst)[i] = o;
}

// ---------------------------------------------------------------------------
// Kernel 2: fp32 [R x Cc] -> bf16 transposed [Cc x R] (tiled through LDS)
// ---------------------------------------------------------------------------
__global__ void __launch_bounds__(256) transpose_cast(const float* __restrict__ src,
                                                      __bf16* __restrict__ dst,
                                                      int R, int Cc) {
  __shared__ float tile[32][33];
  int x = blockIdx.x * 32 + threadIdx.x;
  #pragma unroll
  for (int i = 0; i < 32; i += 8) {
    int y = blockIdx.y * 32 + threadIdx.y + i;
    if (x < Cc && y < R) tile[threadIdx.y + i][threadIdx.x] = src[y * Cc + x];
  }
  __syncthreads();
  int xo = blockIdx.y * 32 + threadIdx.x;          // k index
  #pragma unroll
  for (int i = 0; i < 32; i += 8) {
    int yo = blockIdx.x * 32 + threadIdx.y + i;    // n index
    if (xo < R && yo < Cc) dst[yo * R + xo] = (__bf16)tile[threadIdx.x][threadIdx.y + i];
  }
}

// ---------------------------------------------------------------------------
// 64-wide k-slot: 2 A-frags + 8 B-frags (80 VGPRs).  Two slots double-buffer
// the 16Mx64N GEMM core; all indices compile-time constant.
// ---------------------------------------------------------------------------
struct KSlot { AF a[2]; AF b[2][4]; };

static __device__ __forceinline__ void load_slot(KSlot& s,
                                                 const __bf16* __restrict__ arow,
                                                 const __bf16* const browv[4],
                                                 int kk) {
  #pragma unroll
  for (int u = 0; u < 2; u++) {
    s.a[u] = load_afrag(arow + kk + u * 32);
    #pragma unroll
    for (int i = 0; i < 4; i++) s.b[u][i] = load_bfrag(browv[i] + kk + u * 32);
  }
}

static __device__ __forceinline__ void consume_slot(const KSlot& s, CF acc[4]) {
  #pragma unroll
  for (int u = 0; u < 2; u++)
    #pragma unroll
    for (int i = 0; i < 4; i++)
      acc[i].v = wmma_bf16(s.a[u].v, s.b[u][i].v, acc[i].v);
}

static __device__ __forceinline__ void gemm_core_16x64(const __bf16* __restrict__ arow,
                                                       const __bf16* __restrict__ brow0,
                                                       const __bf16* __restrict__ brow1,
                                                       const __bf16* __restrict__ brow2,
                                                       const __bf16* __restrict__ brow3,
                                                       CF acc[4]) {
  const __bf16* browv[4] = {brow0, brow1, brow2, brow3};
  KSlot s0, s1;
  load_slot(s0, arow, browv, 0);
  // Steady state (single-block body so the sched groups apply): per 128-wide
  // k step, force [20 loads][8 wmma][20 loads][8 wmma] so slot i+1's loads
  // issue a full WMMA-group before slot i is consumed.
  for (int kk = 0; kk < C - 128; kk += 128) {
    load_slot(s1, arow, browv, kk + 64);
    consume_slot(s0, acc);
    load_slot(s0, arow, browv, kk + 128);
    consume_slot(s1, acc);
    __builtin_amdgcn_sched_group_barrier(SG_VMEM_READ, 20, 0);
    __builtin_amdgcn_sched_group_barrier(SG_WMMA,       8, 0);
    __builtin_amdgcn_sched_group_barrier(SG_VMEM_READ, 20, 0);
    __builtin_amdgcn_sched_group_barrier(SG_WMMA,       8, 0);
  }
  // Tail: kk = C-128 (keeps the loop body branch-free)
  load_slot(s1, arow, browv, C - 64);
  consume_slot(s0, acc);
  consume_slot(s1, acc);
}

// ---------------------------------------------------------------------------
// Kernel 3: QKV GEMM.  qkv[M,6144] = x_bf16 @ w_qkvT.  Epilogue scatters
// into q[B,H,T,D] (pre-scaled), k[B,H,T,D], vT[B,H,D,T] bf16.
// ---------------------------------------------------------------------------
__global__ __attribute__((amdgpu_waves_per_eu(1, 2))) void
__launch_bounds__(256) gemm_qkv_kernel(const __bf16* __restrict__ A,
                                       const __bf16* __restrict__ Bt,
                                       __bf16* __restrict__ qb,
                                       __bf16* __restrict__ kb,
                                       __bf16* __restrict__ vtb) {
  const int lane = threadIdx.x & 31;
  const int w    = (blockIdx.x * blockDim.x + threadIdx.x) >> 5;
  const int nTiles = N1 / 64;            // 96
  const int mTile  = w / nTiles;
  const int nTile  = w % nTiles;
  if (mTile >= M / 16) return;
  const int lm = lane & 15, hi = lane >> 4;

  const __bf16* arow  = A + (size_t)(mTile * 16 + lm) * C + 8 * hi;
  const __bf16* bbase = Bt + (size_t)(nTile * 64 + lm) * C + 8 * hi;

  CF acc[4];
  #pragma unroll
  for (int i = 0; i < 4; i++)
    #pragma unroll
    for (int r = 0; r < 8; r++) acc[i].e[r] = 0.f;

  gemm_core_16x64(arow, bbase, bbase + 16 * C, bbase + 32 * C, bbase + 48 * C, acc);

  #pragma unroll
  for (int i = 0; i < 4; i++) {
    const int n     = nTile * 64 + i * 16 + lm;
    const int which = n >> 11;          // 0=q 1=k 2=v (64-tiles never straddle)
    const int cc    = n & 2047;
    const int h     = cc >> 7;
    const int d     = cc & 127;
    #pragma unroll
    for (int r = 0; r < 8; r++) {
      const int gm = mTile * 16 + r + 8 * hi;
      const int b  = gm >> 11;
      const int t  = gm & 2047;
      const float v = acc[i].e[r];
      if (which == 0)      qb[(size_t)((b * H + h) * T + t) * Dh + d] = (__bf16)(v * QSCALE);
      else if (which == 1) kb[(size_t)((b * H + h) * T + t) * Dh + d] = (__bf16)v;
      else                 vtb[(size_t)((b * H + h) * Dh + d) * T + t] = (__bf16)v;
    }
  }
}

// ---------------------------------------------------------------------------
// Kernel 4: causal flash attention.  One wave = 16 queries x 128 head dim.
// V-fragment loads issue before the softmax block so its VALU/transcendental
// work hides their latency; sched groups pin the per-iteration pipeline.
// ---------------------------------------------------------------------------
__global__ __attribute__((amdgpu_waves_per_eu(1, 2))) void
__launch_bounds__(128) attn_kernel(const __bf16* __restrict__ qb,
                                   const __bf16* __restrict__ kb,
                                   const __bf16* __restrict__ vtb,
                                   __bf16* __restrict__ yb) {
  __shared__ __align__(16) __bf16 sP[4][16][40];   // [wave][row][key], padded
  const int lane  = threadIdx.x & 31;
  const int wslot = threadIdx.x >> 5;
  const int w     = (blockIdx.x * blockDim.x + threadIdx.x) >> 5;
  const int qt = w & 127;
  const int h  = (w >> 7) & 15;
  const int b  = w >> 11;
  if (b >= Bdim) return;
  const int lm = lane & 15, hi = lane >> 4;
  const int qbase = qt * 16;

  const __bf16* qhead = qb  + (size_t)(b * H + h) * T * Dh;
  const __bf16* khead = kb  + (size_t)(b * H + h) * T * Dh;
  const __bf16* vhead = vtb + (size_t)(b * H + h) * Dh * T;

  AF aQ[4];
  #pragma unroll
  for (int kc = 0; kc < 4; kc++)
    aQ[kc] = load_afrag(qhead + (size_t)(qbase + lm) * Dh + kc * 32 + 8 * hi);

  CF o[8];
  #pragma unroll
  for (int j = 0; j < 8; j++)
    #pragma unroll
    for (int r = 0; r < 8; r++) o[j].e[r] = 0.f;

  float mrow[8], lrow[8];
  #pragma unroll
  for (int r = 0; r < 8; r++) { mrow[r] = -1e30f; lrow[r] = 0.f; }

  for (int kbase = 0; kbase <= qbase + 15; kbase += 32) {
    // ---- 16 K B-frag loads, then 8 WMMAs (S = Q @ K^T, 32 keys) ----
    AF bk[8];
    #pragma unroll
    for (int kc = 0; kc < 4; kc++) {
      const __bf16* k0 = khead + (size_t)(kbase + lm) * Dh + kc * 32 + 8 * hi;
      bk[kc]     = load_bfrag(k0);
      bk[4 + kc] = load_bfrag(k0 + 16 * Dh);
    }
    CF s0, s1;
    #pragma unroll
    for (int r = 0; r < 8; r++) { s0.e[r] = 0.f; s1.e[r] = 0.f; }
    #pragma unroll
    for (int kc = 0; kc < 4; kc++) {
      s0.v = wmma_bf16(aQ[kc].v, bk[kc].v, s0.v);
      s1.v = wmma_bf16(aQ[kc].v, bk[4 + kc].v, s1.v);
    }

    // ---- V B-frag loads issued early: softmax below hides their latency ----
    AF bv[8];
    #pragma unroll
    for (int j = 0; j < 8; j++)
      bv[j] = load_bfrag(vhead + (size_t)(j * 16 + lm) * T + kbase + 8 * hi);

    // ---- causal mask + online softmax ----
    #pragma unroll
    for (int r = 0; r < 8; r++) {
      const int qrow = qbase + r + 8 * hi;
      const int k0i  = kbase + lm;
      float e0 = (k0i      <= qrow) ? s0.e[r] : -1e30f;
      float e1 = (k0i + 16 <= qrow) ? s1.e[r] : -1e30f;
      float rmax = fmaxf(e0, e1);
      #pragma unroll
      for (int off = 8; off >= 1; off >>= 1)
        rmax = fmaxf(rmax, __shfl_xor(rmax, off, 32));
      const float nm    = fmaxf(mrow[r], rmax);
      const float alpha = __expf(mrow[r] - nm);
      const float p0 = __expf(e0 - nm);
      const float p1 = __expf(e1 - nm);
      float rs = p0 + p1;
      #pragma unroll
      for (int off = 8; off >= 1; off >>= 1)
        rs += __shfl_xor(rs, off, 32);
      mrow[r] = nm;
      lrow[r] = lrow[r] * alpha + rs;
      #pragma unroll
      for (int j = 0; j < 8; j++) o[j].e[r] *= alpha;
      const int row = r + 8 * hi;
      sP[wslot][row][lm]      = (__bf16)p0;
      sP[wslot][row][16 + lm] = (__bf16)p1;
    }

    // ---- P C-frag -> A-frag through per-wave LDS scratch ----
    AF pA;
    pA.q[0] = *reinterpret_cast<const u32x4*>(&sP[wslot][lm][8 * hi]);
    pA.q[1] = *reinterpret_cast<const u32x4*>(&sP[wslot][lm][16 + 8 * hi]);

    // ---- 8 WMMAs (O += P @ V) ----
    #pragma unroll
    for (int j = 0; j < 8; j++)
      o[j].v = wmma_bf16(pA.v, bv[j].v, o[j].v);

    // pipeline description for this iteration
    __builtin_amdgcn_sched_group_barrier(SG_VMEM_READ, 16, 0); // K frags
    __builtin_amdgcn_sched_group_barrier(SG_WMMA,       8, 0); // S = QK^T
    __builtin_amdgcn_sched_group_barrier(SG_VMEM_READ, 16, 0); // V frags
    __builtin_amdgcn_sched_group_barrier(SG_DS_WRITE,  16, 0); // P -> LDS
    __builtin_amdgcn_sched_group_barrier(SG_DS_READ,    2, 0); // P A-frag
    __builtin_amdgcn_sched_group_barrier(SG_WMMA,       8, 0); // O += P@V
  }

  // ---- normalize, store y bf16 [B,T,C] ----
  #pragma unroll
  for (int r = 0; r < 8; r++) {
    const float inv = 1.0f / lrow[r];
    const int t = qbase + r + 8 * hi;
    #pragma unroll
    for (int j = 0; j < 8; j++)
      yb[(size_t)(b * T + t) * C + h * Dh + j * 16 + lm] = (__bf16)(o[j].e[r] * inv);
  }
}

// ---------------------------------------------------------------------------
// Kernel 5: output projection.  out[M,2048] f32 = y_bf16 @ w_outT.
// ---------------------------------------------------------------------------
__global__ __attribute__((amdgpu_waves_per_eu(1, 2))) void
__launch_bounds__(256) gemm_out_kernel(const __bf16* __restrict__ A,
                                       const __bf16* __restrict__ Bt,
                                       float* __restrict__ out) {
  const int lane = threadIdx.x & 31;
  const int w    = (blockIdx.x * blockDim.x + threadIdx.x) >> 5;
  const int nTiles = C / 64;            // 32
  const int mTile  = w / nTiles;
  const int nTile  = w % nTiles;
  if (mTile >= M / 16) return;
  const int lm = lane & 15, hi = lane >> 4;

  const __bf16* arow  = A + (size_t)(mTile * 16 + lm) * C + 8 * hi;
  const __bf16* bbase = Bt + (size_t)(nTile * 64 + lm) * C + 8 * hi;

  CF acc[4];
  #pragma unroll
  for (int i = 0; i < 4; i++)
    #pragma unroll
    for (int r = 0; r < 8; r++) acc[i].e[r] = 0.f;

  gemm_core_16x64(arow, bbase, bbase + 16 * C, bbase + 32 * C, bbase + 48 * C, acc);

  #pragma unroll
  for (int i = 0; i < 4; i++) {
    const int n = nTile * 64 + i * 16 + lm;
    #pragma unroll
    for (int r = 0; r < 8; r++) {
      const int gm = mTile * 16 + r + 8 * hi;
      out[(size_t)gm * C + n] = acc[i].e[r];
    }
  }
}

// ---------------------------------------------------------------------------
// Host-side launcher (graph-capture safe: only kernel launches on `stream`)
// ---------------------------------------------------------------------------
extern "C" void kernel_launch(void* const* d_in, const int* in_sizes, int n_in,
                              void* d_out, int out_size, void* d_ws, size_t ws_size,
                              hipStream_t stream) {
  (void)in_sizes; (void)n_in; (void)out_size; (void)ws_size;
  const float* x     = (const float*)d_in[0];   // [2,2048,2048]
  const float* w_qkv = (const float*)d_in[1];   // [2048,6144]
  const float* w_out = (const float*)d_in[2];   // [2048,2048]
  float* out = (float*)d_out;                   // [2,2048,2048]

  char* ws = (char*)d_ws;
  __bf16* xbf   = (__bf16*)(ws + 0);          // 16 MiB  x as bf16; reused as y
  __bf16* wqkvT = (__bf16*)(ws + 16777216);   // 24 MiB  [6144,2048]
  __bf16* woutT = (__bf16*)(ws + 41943040);   //  8 MiB  [2048,2048]
  __bf16* qb    = (__bf16*)(ws + 50331648);   // 16 MiB  [B,H,T,D] (pre-scaled)
  __bf16* kb    = (__bf16*)(ws + 67108864);   // 16 MiB  [B,H,T,D]
  __bf16* vtb   = (__bf16*)(ws + 83886080);   // 16 MiB  [B,H,D,T]
  __bf16* yb    = xbf;                        // alias: x dead after QKV GEMM

  {
    int n4 = (M * C) / 4;   // 2097152
    cast_f32_bf16<<<n4 / 256, 256, 0, stream>>>(x, xbf, n4);
  }
  transpose_cast<<<dim3(N1 / 32, C / 32), dim3(32, 8), 0, stream>>>(w_qkv, wqkvT, C, N1);
  transpose_cast<<<dim3(C / 32, C / 32), dim3(32, 8), 0, stream>>>(w_out, woutT, C, C);
  gemm_qkv_kernel<<<(M / 16) * (N1 / 64) / 8, 256, 0, stream>>>(xbf, wqkvT, qb, kb, vtb);
  attn_kernel<<<(Bdim * H * (T / 16)) / 4, 128, 0, stream>>>(qb, kb, vtb, yb);
  gemm_out_kernel<<<(M / 16) * (C / 64) / 8, 256, 0, stream>>>(yb, woutT, out);
}
